// MoELayer_79285096284332
// MI455X (gfx1250) — matrix operand order repaired and checked by reading, not compiled, not run on hardware
//
#include <hip/hip_runtime.h>

// ---------------- problem constants ----------------
#define T_TOK   8192
#define DIM     4096
#define NEXP    8
#define TOPK    2
#define HEXP    1024
#define SH      2048
#define TK      (T_TOK * TOPK)

// ---------------- GEMM tile config -----------------
#define BM 128
#define BN 64
#define BK 64
#define LDT 72            // LDS row stride in halves (pad keeps 16B alignment)
#define NTHR 256          // 8 waves of 32

typedef __bf16 bf16_t;
typedef bf16_t v16bf __attribute__((ext_vector_type(16)));
typedef float  v8f   __attribute__((ext_vector_type(8)));
typedef unsigned int u32x4 __attribute__((ext_vector_type(4)));
typedef unsigned int u32x2 __attribute__((ext_vector_type(2)));

// ---------------- helpers ----------------
__device__ __forceinline__ unsigned short f2bf(float f) {
  unsigned u = __builtin_bit_cast(unsigned, f);
  u += 0x7fffu + ((u >> 16) & 1u);          // round-to-nearest-even
  return (unsigned short)(u >> 16);
}
__device__ __forceinline__ unsigned pk2bf(float a, float b) {
  return (unsigned)f2bf(a) | ((unsigned)f2bf(b) << 16);
}
__device__ __forceinline__ float silu(float x) { return x / (1.f + __expf(-x)); }
// CDNA5 prefetch path: lowers to global_prefetch_b8 (speculative, cacheline granular)
__device__ __forceinline__ void pf(const void* p) { __builtin_prefetch(p, 0, 1); }

// A fragment: 16x32 bf16, lane L holds row (L&15); K chunks per ISA layout:
// halves [kh*8,+8) and [16+kh*8,+8) where kh = L>>4.  Two ds_load_b128.
__device__ __forceinline__ v16bf fragA(const unsigned short* A, int msub, int kblk) {
  const int lane = threadIdx.x & 31;
  const unsigned short* p = A + (msub + (lane & 15)) * LDT + kblk + ((lane >> 4) << 3);
  union { u32x4 q[2]; v16bf v; } u;
  u.q[0] = *(const u32x4*)p;
  u.q[1] = *(const u32x4*)(p + 16);
  return u.v;
}
// B fragment: 32x16 bf16 stored [n][k] in LDS; lane L holds col (L&15),
// K halves [kh*8,+8) and [16+kh*8,+8) where kh = L>>4.  Two ds_load_b128.
__device__ __forceinline__ v16bf fragB(const unsigned short* B, int nsub, int kblk) {
  const int lane = threadIdx.x & 31;
  const unsigned short* p = B + (nsub + (lane & 15)) * LDT + kblk + ((lane >> 4) << 3);
  union { u32x4 q[2]; v16bf v; } u;
  u.q[0] = *(const u32x4*)p;
  u.q[1] = *(const u32x4*)(p + 16);
  return u.v;
}
__device__ __forceinline__ v8f wmma_bf16(v16bf a, v16bf b, v8f c) {
  return __builtin_amdgcn_wmma_f32_16x16x32_bf16(false, a, false, b, (short)0, c,
                                                 false, false);
}

// ---------------- kernel 0: x -> bf16, zero counters ----------------
__global__ __launch_bounds__(256) void k_convert(const float* __restrict__ x,
                                                 unsigned short* __restrict__ xb,
                                                 int* __restrict__ cnt) {
  size_t i = ((size_t)blockIdx.x * 256 + threadIdx.x) * 8;
  float4 a = *(const float4*)(x + i);
  float4 b = *(const float4*)(x + i + 4);
  u32x4 o;
  o.x = pk2bf(a.x, a.y); o.y = pk2bf(a.z, a.w);
  o.z = pk2bf(b.x, b.y); o.w = pk2bf(b.z, b.w);
  *(u32x4*)(xb + i) = o;
  if (blockIdx.x == 0 && threadIdx.x < 25) cnt[threadIdx.x] = 0; // counts+cursor+offsets
}

// ---------------- kernel 1: router, top-2, softmax, counts ----------------
__global__ __launch_bounds__(256) void k_router(const float* __restrict__ x,
                                                const float* __restrict__ rw,
                                                int* __restrict__ topk_idx,
                                                float* __restrict__ topk_w,
                                                int* __restrict__ counts) {
  const int wid = threadIdx.x >> 5, lane = threadIdx.x & 31;
  const int t = blockIdx.x * 8 + wid;
  const float4* xr = (const float4*)(x + (size_t)t * DIM);
  float acc[NEXP];
#pragma unroll
  for (int e = 0; e < NEXP; ++e) acc[e] = 0.f;
  for (int i = 0; i < DIM / (4 * 32); ++i) {
    float4 xv = xr[i * 32 + lane];
#pragma unroll
    for (int e = 0; e < NEXP; ++e) {
      float4 wv = *(const float4*)(rw + (size_t)e * DIM + (size_t)(i * 32 + lane) * 4);
      acc[e] += xv.x * wv.x + xv.y * wv.y + xv.z * wv.z + xv.w * wv.w;
    }
  }
#pragma unroll
  for (int e = 0; e < NEXP; ++e)
    for (int o = 16; o > 0; o >>= 1) acc[e] += __shfl_xor(acc[e], o, 32);
  if (lane == 0) {
    int i0 = 0; float v0 = acc[0];
#pragma unroll
    for (int e = 1; e < NEXP; ++e) if (acc[e] > v0) { v0 = acc[e]; i0 = e; }
    int i1 = -1; float v1 = -3.4e38f;
#pragma unroll
    for (int e = 0; e < NEXP; ++e)
      if (e != i0 && acc[e] > v1) { v1 = acc[e]; i1 = e; }
    float s = __expf(v1 - v0);
    float w0 = 1.f / (1.f + s);
    topk_idx[t * 2] = i0; topk_idx[t * 2 + 1] = i1;
    topk_w[t * 2] = w0;   topk_w[t * 2 + 1] = 1.f - w0;
    atomicAdd(&counts[i0], 1);
    atomicAdd(&counts[i1], 1);
  }
}

// ---------------- kernel 2: prefix sum of 8 counts ----------------
__global__ void k_offsets(const int* __restrict__ counts, int* __restrict__ offs) {
  if (threadIdx.x == 0) {
    int s = 0;
    for (int e = 0; e < NEXP; ++e) { offs[e] = s; s += counts[e]; }
    offs[NEXP] = s;
  }
}

// ---------------- kernel 3: scatter tokens into per-expert slots ----------------
__global__ __launch_bounds__(256) void k_scatter(const int* __restrict__ topk_idx,
                                                 const float* __restrict__ topk_w,
                                                 const int* __restrict__ offs,
                                                 int* __restrict__ cursor,
                                                 int* __restrict__ token_map,
                                                 float* __restrict__ weight_map) {
  int t = blockIdx.x * 256 + threadIdx.x;
#pragma unroll
  for (int k = 0; k < TOPK; ++k) {
    int e = topk_idx[t * 2 + k];
    int pos = atomicAdd(&cursor[e], 1);
    int slot = offs[e] + pos;
    token_map[slot] = t;
    weight_map[slot] = topk_w[t * 2 + k];
  }
}

// ============ shared FFN: h = silu(x@w1^T) * (x@w3^T), fused dual-B GEMM ============
__global__ __launch_bounds__(NTHR) void k_shared_up(const unsigned short* __restrict__ xb,
                                                    const float* __restrict__ w1,
                                                    const float* __restrict__ w3,
                                                    unsigned short* __restrict__ hs) {
  __shared__ unsigned short As[BM * LDT];
  __shared__ unsigned short B1s[BN * LDT];
  __shared__ unsigned short B2s[BN * LDT];
  const int tid = threadIdx.x;
  const int t0 = blockIdx.y * BM, n0 = blockIdx.x * BN;
  const int wid = tid >> 5;
  const int wm = (wid & 3) * 32, wn = (wid >> 2) * 32;
  v8f z = {0, 0, 0, 0, 0, 0, 0, 0};
  v8f a1[2][2] = {{z, z}, {z, z}}, a2[2][2] = {{z, z}, {z, z}};

  for (int kb = 0; kb < DIM; kb += BK) {
#pragma unroll
    for (int j = 0; j < 4; ++j) {            // A stage: 128 rows x 64 halves
      int c = tid + j * 256, row = c >> 3, c8 = c & 7;
      *(u32x4*)(As + row * LDT + c8 * 8) =
          *(const u32x4*)(xb + (size_t)(t0 + row) * DIM + kb + c8 * 8);
    }
#pragma unroll
    for (int j = 0; j < 4; ++j) {            // B stage: w1/w3 rows are [n][k] -> copy
      int c = tid + j * 256, n = c >> 4, c4 = c & 15;
      float4 a = *(const float4*)(w1 + (size_t)(n0 + n) * DIM + kb + c4 * 4);
      float4 b = *(const float4*)(w3 + (size_t)(n0 + n) * DIM + kb + c4 * 4);
      u32x2 pa; pa.x = pk2bf(a.x, a.y); pa.y = pk2bf(a.z, a.w);
      u32x2 pb; pb.x = pk2bf(b.x, b.y); pb.y = pk2bf(b.z, b.w);
      *(u32x2*)(B1s + n * LDT + c4 * 4) = pa;
      *(u32x2*)(B2s + n * LDT + c4 * 4) = pb;
    }
    if (kb + BK < DIM) {                     // prefetch next k-panel (global_prefetch_b8)
      int c = tid, n = c >> 4, c4 = c & 15;
      pf(w1 + (size_t)(n0 + n) * DIM + kb + BK + c4 * 4);
      pf(w3 + (size_t)(n0 + n) * DIM + kb + BK + c4 * 4);
      int row = c >> 3, c8 = c & 7;
      pf(xb + (size_t)(t0 + row) * DIM + kb + BK + c8 * 8);
    }
    __syncthreads();
#pragma unroll
    for (int ks = 0; ks < BK; ks += 32) {
      v16bf fa0 = fragA(As, wm, ks), fa1 = fragA(As, wm + 16, ks);
      v16bf fb10 = fragB(B1s, wn, ks), fb11 = fragB(B1s, wn + 16, ks);
      v16bf fb20 = fragB(B2s, wn, ks), fb21 = fragB(B2s, wn + 16, ks);
      a1[0][0] = wmma_bf16(fa0, fb10, a1[0][0]);
      a1[0][1] = wmma_bf16(fa0, fb11, a1[0][1]);
      a1[1][0] = wmma_bf16(fa1, fb10, a1[1][0]);
      a1[1][1] = wmma_bf16(fa1, fb11, a1[1][1]);
      a2[0][0] = wmma_bf16(fa0, fb20, a2[0][0]);
      a2[0][1] = wmma_bf16(fa0, fb21, a2[0][1]);
      a2[1][0] = wmma_bf16(fa1, fb20, a2[1][0]);
      a2[1][1] = wmma_bf16(fa1, fb21, a2[1][1]);
    }
    __syncthreads();
  }
  const int lane = tid & 31, nl = lane & 15, mh = (lane >> 4) * 8;
#pragma unroll
  for (int i = 0; i < 2; ++i)
#pragma unroll
    for (int j = 0; j < 2; ++j)
#pragma unroll
      for (int r = 0; r < 8; ++r) {
        int m = t0 + wm + i * 16 + mh + r;
        int n = n0 + wn + j * 16 + nl;
        float h = silu(a1[i][j][r]) * a2[i][j][r];
        hs[(size_t)m * SH + n] = f2bf(h);
      }
}

// ============ shared down: out = h @ w2^T (writes d_out) ============
__global__ __launch_bounds__(NTHR) void k_shared_down(const unsigned short* __restrict__ hs,
                                                      const float* __restrict__ w2,
                                                      float* __restrict__ out) {
  __shared__ unsigned short As[BM * LDT];
  __shared__ unsigned short Bs[BN * LDT];
  const int tid = threadIdx.x;
  const int t0 = blockIdx.y * BM, n0 = blockIdx.x * BN;
  const int wid = tid >> 5;
  const int wm = (wid & 3) * 32, wn = (wid >> 2) * 32;
  v8f z = {0, 0, 0, 0, 0, 0, 0, 0};
  v8f acc[2][2] = {{z, z}, {z, z}};

  for (int kb = 0; kb < SH; kb += BK) {
#pragma unroll
    for (int j = 0; j < 4; ++j) {
      int c = tid + j * 256, row = c >> 3, c8 = c & 7;
      *(u32x4*)(As + row * LDT + c8 * 8) =
          *(const u32x4*)(hs + (size_t)(t0 + row) * SH + kb + c8 * 8);
    }
#pragma unroll
    for (int j = 0; j < 4; ++j) {            // w2 is [DIM, SH] = [n][k] -> row copy
      int c = tid + j * 256, n = c >> 4, c4 = c & 15;
      float4 a = *(const float4*)(w2 + (size_t)(n0 + n) * SH + kb + c4 * 4);
      u32x2 pa; pa.x = pk2bf(a.x, a.y); pa.y = pk2bf(a.z, a.w);
      *(u32x2*)(Bs + n * LDT + c4 * 4) = pa;
    }
    if (kb + BK < SH) {                      // prefetch next k-panel
      int c = tid, n = c >> 4, c4 = c & 15;
      pf(w2 + (size_t)(n0 + n) * SH + kb + BK + c4 * 4);
      int row = c >> 3, c8 = c & 7;
      pf(hs + (size_t)(t0 + row) * SH + kb + BK + c8 * 8);
    }
    __syncthreads();
#pragma unroll
    for (int ks = 0; ks < BK; ks += 32) {
      v16bf fa0 = fragA(As, wm, ks), fa1 = fragA(As, wm + 16, ks);
      v16bf fb0 = fragB(Bs, wn, ks), fb1 = fragB(Bs, wn + 16, ks);
      acc[0][0] = wmma_bf16(fa0, fb0, acc[0][0]);
      acc[0][1] = wmma_bf16(fa0, fb1, acc[0][1]);
      acc[1][0] = wmma_bf16(fa1, fb0, acc[1][0]);
      acc[1][1] = wmma_bf16(fa1, fb1, acc[1][1]);
    }
    __syncthreads();
  }
  const int lane = tid & 31, nl = lane & 15, mh = (lane >> 4) * 8;
#pragma unroll
  for (int i = 0; i < 2; ++i)
#pragma unroll
    for (int j = 0; j < 2; ++j)
#pragma unroll
      for (int r = 0; r < 8; ++r) {
        int m = t0 + wm + i * 16 + mh + r;
        int n = n0 + wn + j * 16 + nl;
        out[(size_t)m * DIM + n] = acc[i][j][r];
      }
}

// ============ MoE up: grouped GEMM, gathered rows, gate&up fused + SwiGLU ============
__global__ __launch_bounds__(NTHR) void k_moe_up(const unsigned short* __restrict__ xb,
                                                 const float* __restrict__ gate_w,
                                                 const float* __restrict__ up_w,
                                                 unsigned short* __restrict__ hm,
                                                 const int* __restrict__ token_map,
                                                 const int* __restrict__ counts,
                                                 const int* __restrict__ offs) {
  const int e = blockIdx.z;
  const int count = counts[e];
  const int mbase = blockIdx.y * BM;
  if (mbase >= count) return;
  const int slotbase = offs[e];
  const float* gp = gate_w + (size_t)e * DIM * HEXP;
  const float* up = up_w + (size_t)e * DIM * HEXP;

  __shared__ unsigned short As[BM * LDT];
  __shared__ unsigned short B1s[BN * LDT];
  __shared__ unsigned short B2s[BN * LDT];
  __shared__ int smtok[BM];
  const int tid = threadIdx.x;
  const int n0 = blockIdx.x * BN;
  const int wid = tid >> 5;
  const int wm = (wid & 3) * 32, wn = (wid >> 2) * 32;

  if (tid < BM) {
    int gm = mbase + tid;
    smtok[tid] = (gm < count) ? token_map[slotbase + gm] : -1;
  }
  __syncthreads();

  v8f z = {0, 0, 0, 0, 0, 0, 0, 0};
  v8f a1[2][2] = {{z, z}, {z, z}}, a2[2][2] = {{z, z}, {z, z}};
  u32x4 zero4 = {0, 0, 0, 0};

  for (int kb = 0; kb < DIM; kb += BK) {
#pragma unroll
    for (int j = 0; j < 4; ++j) {            // gathered A rows
      int c = tid + j * 256, row = c >> 3, c8 = c & 7;
      int tok = smtok[row];
      u32x4 v = zero4;
      if (tok >= 0) v = *(const u32x4*)(xb + (size_t)tok * DIM + kb + c8 * 8);
      *(u32x4*)(As + row * LDT + c8 * 8) = v;
    }
#pragma unroll
    for (int j = 0; j < 4; ++j) {            // B: [k][n] in memory -> transpose to [n][k]
      int c = tid + j * 256, k = c >> 4, n4 = c & 15;
      float4 g = *(const float4*)(gp + (size_t)(kb + k) * HEXP + n0 + n4 * 4);
      float4 u = *(const float4*)(up + (size_t)(kb + k) * HEXP + n0 + n4 * 4);
      B1s[(n4 * 4 + 0) * LDT + k] = f2bf(g.x);
      B1s[(n4 * 4 + 1) * LDT + k] = f2bf(g.y);
      B1s[(n4 * 4 + 2) * LDT + k] = f2bf(g.z);
      B1s[(n4 * 4 + 3) * LDT + k] = f2bf(g.w);
      B2s[(n4 * 4 + 0) * LDT + k] = f2bf(u.x);
      B2s[(n4 * 4 + 1) * LDT + k] = f2bf(u.y);
      B2s[(n4 * 4 + 2) * LDT + k] = f2bf(u.z);
      B2s[(n4 * 4 + 3) * LDT + k] = f2bf(u.w);
    }
    if (kb + BK < DIM) {                     // prefetch next weight k-panel
      int c = tid, k = c >> 4, n4 = c & 15;
      pf(gp + (size_t)(kb + BK + k) * HEXP + n0 + n4 * 4);
      pf(up + (size_t)(kb + BK + k) * HEXP + n0 + n4 * 4);
    }
    __syncthreads();
#pragma unroll
    for (int ks = 0; ks < BK; ks += 32) {
      v16bf fa0 = fragA(As, wm, ks), fa1 = fragA(As, wm + 16, ks);
      v16bf fb10 = fragB(B1s, wn, ks), fb11 = fragB(B1s, wn + 16, ks);
      v16bf fb20 = fragB(B2s, wn, ks), fb21 = fragB(B2s, wn + 16, ks);
      a1[0][0] = wmma_bf16(fa0, fb10, a1[0][0]);
      a1[0][1] = wmma_bf16(fa0, fb11, a1[0][1]);
      a1[1][0] = wmma_bf16(fa1, fb10, a1[1][0]);
      a1[1][1] = wmma_bf16(fa1, fb11, a1[1][1]);
      a2[0][0] = wmma_bf16(fa0, fb20, a2[0][0]);
      a2[0][1] = wmma_bf16(fa0, fb21, a2[0][1]);
      a2[1][0] = wmma_bf16(fa1, fb20, a2[1][0]);
      a2[1][1] = wmma_bf16(fa1, fb21, a2[1][1]);
    }
    __syncthreads();
  }
  const int lane = tid & 31, nl = lane & 15, mh = (lane >> 4) * 8;
#pragma unroll
  for (int i = 0; i < 2; ++i)
#pragma unroll
    for (int j = 0; j < 2; ++j)
#pragma unroll
      for (int r = 0; r < 8; ++r) {
        int rl = wm + i * 16 + mh + r;
        int gm = mbase + rl;
        if (gm < count) {
          int n = n0 + wn + j * 16 + nl;
          float h = silu(a1[i][j][r]) * a2[i][j][r];
          hm[(size_t)(slotbase + gm) * HEXP + n] = f2bf(h);
        }
      }
}

// ============ MoE down: grouped GEMM, scaled atomic scatter-add into d_out ============
__global__ __launch_bounds__(NTHR) void k_moe_down(const unsigned short* __restrict__ hm,
                                                   const float* __restrict__ down_w,
                                                   float* __restrict__ out,
                                                   const int* __restrict__ token_map,
                                                   const float* __restrict__ weight_map,
                                                   const int* __restrict__ counts,
                                                   const int* __restrict__ offs) {
  const int e = blockIdx.z;
  const int count = counts[e];
  const int mbase = blockIdx.y * BM;
  if (mbase >= count) return;
  const int slotbase = offs[e];
  const float* dp = down_w + (size_t)e * HEXP * DIM;

  __shared__ unsigned short As[BM * LDT];
  __shared__ unsigned short Bs[BN * LDT];
  __shared__ int smt[BM];
  __shared__ float smw[BM];
  const int tid = threadIdx.x;
  const int n0 = blockIdx.x * BN;
  const int wid = tid >> 5;
  const int wm = (wid & 3) * 32, wn = (wid >> 2) * 32;

  if (tid < BM) {
    int gm = mbase + tid;
    bool v = gm < count;
    smt[tid] = v ? token_map[slotbase + gm] : -1;
    smw[tid] = v ? weight_map[slotbase + gm] : 0.f;
  }
  __syncthreads();

  v8f z = {0, 0, 0, 0, 0, 0, 0, 0};
  v8f acc[2][2] = {{z, z}, {z, z}};

  for (int kb = 0; kb < HEXP; kb += BK) {
#pragma unroll
    for (int j = 0; j < 4; ++j) {            // A rows: compacted hidden slots
      int c = tid + j * 256, row = c >> 3, c8 = c & 7;
      int slot = slotbase + mbase + row;
      if (slot > TK - 1) slot = TK - 1;      // clamp (stores are guarded anyway)
      *(u32x4*)(As + row * LDT + c8 * 8) =
          *(const u32x4*)(hm + (size_t)slot * HEXP + kb + c8 * 8);
    }
#pragma unroll
    for (int j = 0; j < 4; ++j) {            // B: down_proj [k=H][n=DIM] -> transpose
      int c = tid + j * 256, k = c >> 4, n4 = c & 15;
      float4 g = *(const float4*)(dp + (size_t)(kb + k) * DIM + n0 + n4 * 4);
      Bs[(n4 * 4 + 0) * LDT + k] = f2bf(g.x);
      Bs[(n4 * 4 + 1) * LDT + k] = f2bf(g.y);
      Bs[(n4 * 4 + 2) * LDT + k] = f2bf(g.z);
      Bs[(n4 * 4 + 3) * LDT + k] = f2bf(g.w);
    }
    if (kb + BK < HEXP) {                    // prefetch next weight k-panel
      int c = tid, k = c >> 4, n4 = c & 15;
      pf(dp + (size_t)(kb + BK + k) * DIM + n0 + n4 * 4);
    }
    __syncthreads();
#pragma unroll
    for (int ks = 0; ks < BK; ks += 32) {
      v16bf fa0 = fragA(As, wm, ks), fa1 = fragA(As, wm + 16, ks);
      v16bf fb0 = fragB(Bs, wn, ks), fb1 = fragB(Bs, wn + 16, ks);
      acc[0][0] = wmma_bf16(fa0, fb0, acc[0][0]);
      acc[0][1] = wmma_bf16(fa0, fb1, acc[0][1]);
      acc[1][0] = wmma_bf16(fa1, fb0, acc[1][0]);
      acc[1][1] = wmma_bf16(fa1, fb1, acc[1][1]);
    }
    __syncthreads();
  }
  const int lane = tid & 31, nl = lane & 15, mh = (lane >> 4) * 8;
#pragma unroll
  for (int i = 0; i < 2; ++i)
#pragma unroll
    for (int j = 0; j < 2; ++j)
#pragma unroll
      for (int r = 0; r < 8; ++r) {
        int rl = wm + i * 16 + mh + r;
        int tok = smt[rl];
        if (tok >= 0) {
          int n = n0 + wn + j * 16 + nl;
          atomicAdd(&out[(size_t)tok * DIM + n], acc[i][j][r] * smw[rl]);
        }
      }
}

// ---------------- host-side launch ----------------
extern "C" void kernel_launch(void* const* d_in, const int* in_sizes, int n_in,
                              void* d_out, int out_size, void* d_ws, size_t ws_size,
                              hipStream_t stream) {
  (void)in_sizes; (void)n_in; (void)out_size; (void)ws_size;
  const float* x         = (const float*)d_in[0];
  const float* router_w  = (const float*)d_in[1];
  const float* gate_proj = (const float*)d_in[2];
  const float* up_proj   = (const float*)d_in[3];
  const float* down_proj = (const float*)d_in[4];
  const float* w1        = (const float*)d_in[5];
  const float* w2        = (const float*)d_in[6];
  const float* w3        = (const float*)d_in[7];
  float* out = (float*)d_out;

  // workspace carve (~128.3 MB total)
  char* ws = (char*)d_ws;
  size_t o = 0;
  unsigned short* xb = (unsigned short*)(ws + o); o += (size_t)T_TOK * DIM * 2;   // 64 MB
  unsigned short* hs = (unsigned short*)(ws + o); o += (size_t)T_TOK * SH * 2;    // 32 MB
  unsigned short* hm = (unsigned short*)(ws + o); o += (size_t)TK * HEXP * 2;     // 32 MB
  int*   topk_idx   = (int*)(ws + o);   o += (size_t)T_TOK * 2 * 4;
  float* topk_w     = (float*)(ws + o); o += (size_t)T_TOK * 2 * 4;
  int*   token_map  = (int*)(ws + o);   o += (size_t)TK * 4;
  float* weight_map = (float*)(ws + o); o += (size_t)TK * 4;
  int*   cnt        = (int*)(ws + o);   // [0..7]=counts [8..15]=cursor [16..24]=offsets
  int* counts = cnt;
  int* cursor = cnt + 8;
  int* offs   = cnt + 16;

  // 1) bf16 activations + zero counters
  k_convert<<<(size_t)T_TOK * DIM / (256 * 8), 256, 0, stream>>>(x, xb, cnt);
  // 2) router + top-2 + softmax + expert counts
  k_router<<<T_TOK / 8, 256, 0, stream>>>(x, router_w, topk_idx, topk_w, counts);
  // 3) offsets, 4) compacted scatter (grouped-GEMM dispatch lists)
  k_offsets<<<1, 32, 0, stream>>>(counts, offs);
  k_scatter<<<T_TOK / 256, 256, 0, stream>>>(topk_idx, topk_w, offs, cursor,
                                             token_map, weight_map);
  // 5) shared FFN up (fused SwiGLU) -> bf16 hidden
  k_shared_up<<<dim3(SH / BN, T_TOK / BM), NTHR, 0, stream>>>(xb, w1, w3, hs);
  // 6) shared FFN down: writes d_out
  k_shared_down<<<dim3(DIM / BN, T_TOK / BM), NTHR, 0, stream>>>(hs, w2, out);
  // 7) MoE up (grouped, gathered rows, fused SwiGLU) -> bf16 hidden per slot
  k_moe_up<<<dim3(HEXP / BN, T_TOK / BM, NEXP), NTHR, 0, stream>>>(
      xb, gate_proj, up_proj, hm, token_map, counts, offs);
  // 8) MoE down (grouped) with scaled atomic add into d_out
  k_moe_down<<<dim3(DIM / BN, T_TOK / BM, NEXP), NTHR, 0, stream>>>(
      hm, down_proj, out, token_map, weight_map, counts, offs);
}
